// RelativePositionSelfAttention_4415226380580
// MI455X (gfx1250) — compile-verified, matrix operand order checked by reading, not run on hardware
//
#include <hip/hip_runtime.h>
#include <hip/hip_bf16.h>

// ---------------- config ----------------
#define BATCH 8
#define SEQ   1024
#define HID   128
#define NPOS  129          // 2*64+1 relative positions
#define NPP   160          // NPOS padded to a multiple of 32 (WMMA K granularity)
#define BS    (BATCH*SEQ)  // 8192 rows
#define SCALE 0.08838834764831845f   // 1/sqrt(128)

typedef __attribute__((ext_vector_type(16))) __bf16 v16bf;
typedef __attribute__((ext_vector_type(8)))  float  v8f;

// ---- WMMA bf16 fragment loader ----
// Loads a 16x32 (rows x K) bf16 fragment from a row-major buffer (LDS or global).
// Layout per CDNA5 ISA 7.12.2 (16-bit A-matrix):
//   lane<16 : row = lane,    K = {k0..k0+7} u {k0+16..k0+23}
//   lane>=16: row = lane-16, K = {k0+8..k0+15} u {k0+24..k0+31}
// Used for A (row-major [M][K]) and for B supplied transposed ([N][K] row-major).
__device__ __forceinline__ v16bf ld_frag(const __bf16* buf, int row0, int k0, int ld) {
  const int lane = threadIdx.x & 31;
  const int r    = lane & 15;
  const int kb   = (lane >> 4) << 3;            // 0 or 8
  const __bf16* p = buf + (size_t)(row0 + r) * ld + k0 + kb;
  v16bf v;
  ((uint4*)&v)[0] = *(const uint4*)(p);         // 8 bf16 = 16B, aligned
  ((uint4*)&v)[1] = *(const uint4*)(p + 16);    // next 8 bf16
  return v;
}

#define WMMA_BF16(acc, a, b) \
  acc = __builtin_amdgcn_wmma_f32_16x16x32_bf16(false, (a), false, (b), (short)0, (acc), false, false)

// ============================================================================
// Kernel 1: q,k,v projections.  q,k -> bf16 [BS][HID]; v -> bf16 transposed
// per batch [BATCH][HID][SEQ] so attn@v B-operand is contiguous in K.
// One WG = 32 rows, 8 waves. Wave w owns n-tile w; iterates which(3) x mt(2).
// A-fragments preloaded once per wave (invariant across 'which').
// ============================================================================
__global__ __launch_bounds__(256) void qkv_proj_kernel(
    const float* __restrict__ x,
    const float* __restrict__ Wq, const float* __restrict__ bq,
    const float* __restrict__ Wk, const float* __restrict__ bk,
    const float* __restrict__ Wv, const float* __restrict__ bv,
    __bf16* __restrict__ qb, __bf16* __restrict__ kb, __bf16* __restrict__ vT) {
  __shared__ alignas(16) __bf16 sWt[3][HID][HID];  // weights, transposed [n][k], 96KB
  __shared__ alignas(16) __bf16 sx[32][HID];       // x rows bf16, 8KB
  __shared__ float sbias[3][HID];

  const int t = threadIdx.x;
  const int row0 = blockIdx.x * 32;                // global row = b*SEQ + s

  const float* Ws[3] = {Wq, Wk, Wv};
  const float* bs[3] = {bq, bk, bv};
  for (int w = 0; w < 3; ++w) {
    for (int idx = t; idx < HID * HID; idx += 256) {
      int k = idx >> 7, n = idx & 127;
      sWt[w][n][k] = (__bf16)Ws[w][idx];           // transpose into LDS
    }
    if (t < HID) sbias[w][t] = bs[w][t];
  }
  for (int idx = t; idx < 32 * HID; idx += 256) {
    int r = idx >> 7, c = idx & 127;
    sx[r][c] = (__bf16)x[(size_t)(row0 + r) * HID + c];
  }
  __syncthreads();

  const int wave = t >> 5, lane = t & 31;
  const int nt = wave;                             // fixed n-tile per wave

  // preload invariant A fragments: [mt][ks]
  v16bf amt[2][4];
#pragma unroll
  for (int mt = 0; mt < 2; ++mt)
#pragma unroll
    for (int ks = 0; ks < 4; ++ks)
      amt[mt][ks] = ld_frag(&sx[0][0], mt * 16, ks * 32, HID);

  for (int sel = 0; sel < 6; ++sel) {              // which(3) x mt(2)
    const int which = sel >> 1, mt = sel & 1;
    v8f acc = {};
#pragma unroll
    for (int ks = 0; ks < 4; ++ks) {
      v16bf bv2 = ld_frag(&sWt[which][0][0], nt * 16, ks * 32, HID);
      WMMA_BF16(acc, amt[mt][ks], bv2);
    }
    const int n = lane & 15, mh = (lane >> 4) * 8;
    const int gcol = nt * 16 + n;
    const float bias = sbias[which][gcol];
    if (which < 2) {
      __bf16* dst = (which == 0) ? qb : kb;
#pragma unroll
      for (int r = 0; r < 8; ++r) {
        int grow = row0 + mt * 16 + mh + r;
        dst[(size_t)grow * HID + gcol] = (__bf16)(acc[r] + bias);
      }
    } else {
#pragma unroll
      for (int r = 0; r < 8; ++r) {                // v stored transposed
        int grow = row0 + mt * 16 + mh + r;
        int b = grow >> 10, s = grow & (SEQ - 1);
        vT[((size_t)b * HID + gcol) * SEQ + s] = (__bf16)(acc[r] + bias);
      }
    }
  }
}

// ============================================================================
// Kernel 2: scores = qk^T/sqrt(H) + gather(q@rel_k^T), softmax, attn out (fp32),
// and relative-bucket sums a_rel (bf16, K-padded to NPP).
// One WG = (batch b, 16 query rows). 8 waves. q fragments hoisted (invariant).
// ============================================================================
__global__ __launch_bounds__(256) void attn_scores_kernel(
    const float* __restrict__ rel_k,
    const __bf16* __restrict__ qb, const __bf16* __restrict__ kb,
    float* __restrict__ attn_out, __bf16* __restrict__ arel) {
  __shared__ alignas(16) __bf16 sRk[NPP][HID];   // rel_k row-major (== B^T), 40KB
  __shared__ alignas(16) __bf16 sQ[16][HID];     // 4KB
  __shared__ float sQr[16][NPP];                 // 10KB
  __shared__ float sSc[16][SEQ];                 // 64KB
  __shared__ float sRed[16][16];

  const int t = threadIdx.x, wave = t >> 5, lane = t & 31;
  const int b = blockIdx.x >> 6;                 // 64 query tiles per batch
  const int i0 = (blockIdx.x & 63) * 16;
  const int rowbase = b * SEQ + i0;

  for (int idx = t; idx < NPP * HID; idx += 256) {
    int p = idx >> 7, c = idx & 127;
    sRk[p][c] = (p < NPOS) ? (__bf16)rel_k[p * HID + c] : (__bf16)0.0f;
  }
  for (int idx = t; idx < 16 * HID; idx += 256) {
    int r = idx >> 7, c = idx & 127;
    sQ[r][c] = qb[(size_t)(rowbase + r) * HID + c];
  }
  __syncthreads();

  // hoisted, loop-invariant q A-fragments
  v16bf qf[4];
#pragma unroll
  for (int ks = 0; ks < 4; ++ks) qf[ks] = ld_frag(&sQ[0][0], 0, ks * 32, HID);

  // qr[m][p] = q[m,:] . rel_k[p,:]  (10 n-tiles over padded NPP)
  for (int nt = wave; nt < 10; nt += 8) {
    v8f acc = {};
#pragma unroll
    for (int ks = 0; ks < 4; ++ks) {
      v16bf bv2 = ld_frag(&sRk[0][0], nt * 16, ks * 32, HID);
      WMMA_BF16(acc, qf[ks], bv2);
    }
    const int n = lane & 15, mh = (lane >> 4) * 8;
#pragma unroll
    for (int r = 0; r < 8; ++r) sQr[mh + r][nt * 16 + n] = acc[r];
  }
  __syncthreads();

  // scores: 64 key tiles; k rows stream from global (B^T == k row-major)
  const __bf16* kbat = kb + (size_t)b * SEQ * HID;
  for (int jt = wave; jt < 64; jt += 8) {
    if (jt + 8 < 64)  // prefetch next tile this wave will touch (4KB spread over lanes)
      __builtin_prefetch(kbat + (size_t)(jt + 8) * 16 * HID + (size_t)(t & 31) * 64, 0, 1);
    v8f acc = {};
#pragma unroll
    for (int ks = 0; ks < 4; ++ks) {
      v16bf bv2 = ld_frag(kbat, jt * 16, ks * 32, HID);
      WMMA_BF16(acc, qf[ks], bv2);
    }
    const int n = lane & 15, mh = (lane >> 4) * 8;
    const int j = jt * 16 + n;
#pragma unroll
    for (int r = 0; r < 8; ++r) {
      int m = mh + r;
      int d = j - (i0 + m);
      d = d < -64 ? -64 : (d > 64 ? 64 : d);
      sSc[m][j] = acc[r] * SCALE + sQr[m][d + 64];
    }
  }
  __syncthreads();

  // softmax: 16 threads per row
  const int row = t >> 4, sub = t & 15;
  float mx = -3.0e38f;
  for (int j = sub; j < SEQ; j += 16) mx = fmaxf(mx, sSc[row][j]);
  sRed[row][sub] = mx;
  __syncthreads();
  if (sub == 0) {
    float m2 = sRed[row][0];
    for (int u = 1; u < 16; ++u) m2 = fmaxf(m2, sRed[row][u]);
    sRed[row][0] = m2;
  }
  __syncthreads();
  mx = sRed[row][0];
  float sum = 0.f;
  for (int j = sub; j < SEQ; j += 16) {
    float e = __expf(sSc[row][j] - mx);
    sSc[row][j] = e;
    sum += e;
  }
  __syncthreads();
  sRed[row][sub] = sum;
  __syncthreads();
  if (sub == 0) {
    float s2 = 0.f;
    for (int u = 0; u < 16; ++u) s2 += sRed[row][u];
    sRed[row][0] = 1.0f / s2;
  }
  __syncthreads();
  const float inv = sRed[row][0];
  float* attn_row = attn_out + (size_t)(rowbase + row) * SEQ;
  for (int j = sub; j < SEQ; j += 16) {
    float p = sSc[row][j] * inv;
    sSc[row][j] = p;
    attn_row[j] = p;
  }
  __syncthreads();

  // a_rel bucket sums: interior p -> single diagonal element; p=0/128 -> tails
  for (int idx = t; idx < 16 * NPP; idx += 256) {
    int m = idx / NPP, p = idx % NPP;
    int gi = i0 + m;
    float val = 0.f;
    if (p < NPOS) {
      if (p == 0) {
        int jmax = gi - 64;
        for (int j = 0; j <= jmax; ++j) val += sSc[m][j];
      } else if (p == NPOS - 1) {
        for (int j = gi + 64; j < SEQ; ++j) val += sSc[m][j];
      } else {
        int j = gi + p - 64;
        if (j >= 0 && j < SEQ) val = sSc[m][j];
      }
    }
    arel[(size_t)(rowbase + m) * NPP + p] = (__bf16)val;
  }
}

// ============================================================================
// Kernel 3: ctx = attn@v + a_rel@rel_v; out = ctx@Wo + bo; residual + LayerNorm.
// One WG = 16 rows, 8 waves (one H-tile each).
// ============================================================================
__global__ __launch_bounds__(256) void out_ln_kernel(
    const float* __restrict__ x, const float* __restrict__ rel_v,
    const float* __restrict__ Wo, const float* __restrict__ bo,
    const float* __restrict__ gamma, const float* __restrict__ beta,
    const __bf16* __restrict__ vT, const __bf16* __restrict__ arel,
    const float* __restrict__ attn, float* __restrict__ y) {
  __shared__ alignas(16) __bf16 sA[16][SEQ];     // attn bf16, 32KB
  __shared__ alignas(16) __bf16 sRvT[HID][NPP];  // rel_v transposed [h][p], 40KB
  __shared__ alignas(16) __bf16 sWoT[HID][HID];  // 32KB
  __shared__ alignas(16) __bf16 sCtx[16][HID];   // 4KB
  __shared__ float sY[16][HID];                  // 8KB
  __shared__ float sRed[16][16];

  const int t = threadIdx.x, wave = t >> 5, lane = t & 31;
  const int b = blockIdx.x >> 6;
  const int i0 = (blockIdx.x & 63) * 16;
  const int rowbase = b * SEQ + i0;

  for (int idx = t; idx < 16 * SEQ; idx += 256) {
    int r = idx >> 10, j = idx & (SEQ - 1);
    sA[r][j] = (__bf16)attn[(size_t)(rowbase + r) * SEQ + j];
  }
  for (int idx = t; idx < HID * NPP; idx += 256) {
    int h = idx / NPP, p = idx % NPP;
    sRvT[h][p] = (p < NPOS) ? (__bf16)rel_v[p * HID + h] : (__bf16)0.0f;
  }
  for (int idx = t; idx < HID * HID; idx += 256) {
    int k = idx >> 7, n = idx & 127;
    sWoT[n][k] = (__bf16)Wo[idx];
  }
  __syncthreads();

  {  // ctx tile: n-tile = wave
    const int nt = wave;
    v8f acc = {};
    const __bf16* vTb = vT + (size_t)b * HID * SEQ;  // [h][s] row-major
#pragma unroll 4
    for (int ks = 0; ks < 32; ++ks) {                // K = SEQ
      if (ks + 1 < 32)  // prefetch this lane's row chunk for the next k-step
        __builtin_prefetch(vTb + (size_t)(nt * 16 + (lane & 15)) * SEQ + (ks + 1) * 32, 0, 1);
      v16bf av = ld_frag(&sA[0][0], 0, ks * 32, SEQ);
      v16bf bv2 = ld_frag(vTb, nt * 16, ks * 32, SEQ);
      WMMA_BF16(acc, av, bv2);
    }
    const __bf16* ar = arel + (size_t)rowbase * NPP;
#pragma unroll
    for (int ks = 0; ks < 5; ++ks) {                 // K = NPP (zero padded)
      v16bf av = ld_frag(ar, 0, ks * 32, NPP);
      v16bf bv2 = ld_frag(&sRvT[0][0], nt * 16, ks * 32, NPP);
      WMMA_BF16(acc, av, bv2);
    }
    const int n = lane & 15, mh = (lane >> 4) * 8;
#pragma unroll
    for (int r = 0; r < 8; ++r) sCtx[mh + r][nt * 16 + n] = (__bf16)acc[r];
  }
  __syncthreads();

  {  // output projection + bias + residual
    const int nt = wave;
    v8f acc = {};
#pragma unroll
    for (int ks = 0; ks < 4; ++ks) {
      v16bf av = ld_frag(&sCtx[0][0], 0, ks * 32, HID);
      v16bf bv2 = ld_frag(&sWoT[0][0], nt * 16, ks * 32, HID);
      WMMA_BF16(acc, av, bv2);
    }
    const int n = lane & 15, mh = (lane >> 4) * 8;
    const int gc = nt * 16 + n;
    const float bias = bo[gc];
#pragma unroll
    for (int r = 0; r < 8; ++r) {
      int m = mh + r;
      sY[m][gc] = acc[r] + bias + x[(size_t)(rowbase + m) * HID + gc];
    }
  }
  __syncthreads();

  // LayerNorm over HID, 16 threads per row
  const int row = t >> 4, sub = t & 15;
  float s = 0.f;
  for (int c = sub; c < HID; c += 16) s += sY[row][c];
  sRed[row][sub] = s;
  __syncthreads();
  if (sub == 0) {
    float m = 0.f;
    for (int u = 0; u < 16; ++u) m += sRed[row][u];
    sRed[row][0] = m * (1.0f / HID);
  }
  __syncthreads();
  const float mu = sRed[row][0];
  float vs = 0.f;
  for (int c = sub; c < HID; c += 16) { float d = sY[row][c] - mu; vs += d * d; }
  __syncthreads();
  sRed[row][sub] = vs;
  __syncthreads();
  if (sub == 0) {
    float vv = 0.f;
    for (int u = 0; u < 16; ++u) vv += sRed[row][u];
    sRed[row][0] = rsqrtf(vv * (1.0f / HID) + 1e-5f);
  }
  __syncthreads();
  const float rinv = sRed[row][0];
  for (int c = sub; c < HID; c += 16) {
    y[(size_t)(rowbase + row) * HID + c] = (sY[row][c] - mu) * rinv * gamma[c] + beta[c];
  }
}

// ============================================================================
extern "C" void kernel_launch(void* const* d_in, const int* in_sizes, int n_in,
                              void* d_out, int out_size, void* d_ws, size_t ws_size,
                              hipStream_t stream) {
  const float* x     = (const float*)d_in[0];
  const float* Wq    = (const float*)d_in[1];
  const float* bq    = (const float*)d_in[2];
  const float* Wk    = (const float*)d_in[3];
  const float* bk    = (const float*)d_in[4];
  const float* Wv    = (const float*)d_in[5];
  const float* bv    = (const float*)d_in[6];
  const float* rel_k = (const float*)d_in[7];
  const float* rel_v = (const float*)d_in[8];
  const float* Wo    = (const float*)d_in[9];
  const float* bo    = (const float*)d_in[10];
  const float* gamma = (const float*)d_in[11];
  const float* beta  = (const float*)d_in[12];

  float* y    = (float*)d_out;                 // [BS*HID]
  float* attn = y + (size_t)BS * HID;          // [BATCH*SEQ*SEQ]

  __bf16* qb   = (__bf16*)d_ws;                // [BS][HID]
  __bf16* kbuf = qb + (size_t)BS * HID;        // [BS][HID]
  __bf16* vT   = kbuf + (size_t)BS * HID;      // [BATCH][HID][SEQ]
  __bf16* arel = vT + (size_t)BATCH * HID * SEQ;  // [BS][NPP]

  qkv_proj_kernel<<<BS / 32, 256, 0, stream>>>(x, Wq, bq, Wk, bk, Wv, bv, qb, kbuf, vT);
  attn_scores_kernel<<<BATCH * (SEQ / 16), 256, 0, stream>>>(rel_k, qb, kbuf, attn, arel);
  out_ln_kernel<<<BATCH * (SEQ / 16), 256, 0, stream>>>(x, rel_v, Wo, bo, gamma, beta,
                                                        vT, arel, attn, y);
}
